// QuantLinear_1468878815914
// MI455X (gfx1250) — compile-verified, matrix operand order checked by reading, not run on hardware
//
#include <hip/hip_runtime.h>

typedef __attribute__((ext_vector_type(16))) _Float16 v16h;
typedef __attribute__((ext_vector_type(8)))  _Float16 v8h;
typedef __attribute__((ext_vector_type(2)))  _Float16 h2;
typedef __attribute__((ext_vector_type(8)))  float    v8f;
typedef __attribute__((ext_vector_type(4)))  int      i4;

constexpr int K_DIM   = 8192;
constexpr int N_DIM   = 28672;
constexpr int M_DIM   = 32;
constexpr int GROUP_Q = 128;
constexpr int WAVES   = 4;                         // K-split ways inside a block
constexpr int COLS    = 32;                        // output columns per block
constexpr int K_PER_WAVE   = K_DIM / WAVES;        // 2048
constexpr int GRP_PER_WAVE = K_PER_WAVE / GROUP_Q; // 16
constexpr int CHUNKS       = K_PER_WAVE / 32;      // 64 K-chunks per wave

union BFrag { v16h v; h2  p[8]; };
union AFrag { v16h v; v8h h[2]; };

// One K-chunk worth of per-lane raw data: 4x B128 weights + 4x B128 A halves.
struct Chunk {
    i4  w0lo, w0hi, w1lo, w1hi;
    v8h a0lo, a0hi, a1lo, a1hi;
};

__device__ __forceinline__ Chunk load_chunk(const int* __restrict__ q0,
                                            const int* __restrict__ q1,
                                            const _Float16* __restrict__ a0p,
                                            const _Float16* __restrict__ a1p,
                                            int kb) {
    Chunk c;
    // weights: one int32 = one packed byte = k pair (low nibble = even k).
    // Non-temporal: streamed exactly once; keep A resident in L2.
    c.w0lo = __builtin_nontemporal_load((const i4*)(q0 + (kb >> 1)));
    c.w0hi = __builtin_nontemporal_load((const i4*)(q0 + ((kb + 16) >> 1)));
    c.w1lo = __builtin_nontemporal_load((const i4*)(q1 + (kb >> 1)));
    c.w1hi = __builtin_nontemporal_load((const i4*)(q1 + ((kb + 16) >> 1)));
    c.a0lo = *(const v8h*)(a0p + kb);
    c.a0hi = *(const v8h*)(a0p + kb + 16);
    c.a1lo = *(const v8h*)(a1p + kb);
    c.a1hi = *(const v8h*)(a1p + kb + 16);
    return c;
}

// nibble pair -> two dequantized fp16: 0x6400|w encodes (1024+w) exactly;
// pk_add(-1024) is exact, then one packed FMA applies (w)*s - z*s.
__device__ __forceinline__ h2 dq_pair(unsigned q, h2 s, h2 nzs) {
    unsigned t = (((q << 12) | q) & 0x000F000Fu) | 0x64006400u;
    h2 hv = __builtin_bit_cast(h2, t);
    const h2 k1024 = {(_Float16)1024.0f, (_Float16)1024.0f};
    hv = hv - k1024;
    return hv * s + nzs;           // v_pk_fma_f16
}

// C[32,N] = A[32,K] * dequant(qweight)[N,K]^T + bias
// Block: 128 threads = 4 wave32. All 4 waves cover the SAME 32 columns over
// different K quarters; partials reduced through LDS. Per chunk: 2 B tiles x
// 2 A tiles -> 4x v_wmma_f32_16x16x32_f16. Loads software-pipelined one
// chunk ahead so WMMA issue overlaps the VMEM round trip.
__global__ __launch_bounds__(128) void gptq_w4a16_wmma(
    const _Float16* __restrict__ A,
    const int*      __restrict__ qweight,
    const _Float16* __restrict__ scales,
    const _Float16* __restrict__ zeros,
    const _Float16* __restrict__ bias,
    _Float16*       __restrict__ out)
{
    __shared__ float red[WAVES][M_DIM][COLS];    // 16 KB

    const int lane   = threadIdx.x & 31;
    const int wave   = threadIdx.x >> 5;
    const int lane15 = lane & 15;
    const int khalf  = lane >> 4;                // k-half per ISA 16-bit layout
    const int col0   = blockIdx.x * COLS;
    const int n0     = col0 + lane15;
    const int n1     = n0 + 16;
    const int kbase  = wave * K_PER_WAVE;

    const int* __restrict__ q0 = qweight + (size_t)n0 * (K_DIM / 2) + kbase / 2;
    const int* __restrict__ q1 = qweight + (size_t)n1 * (K_DIM / 2) + kbase / 2;
    const _Float16* __restrict__ a0p = A + (size_t)lane15 * K_DIM + kbase;
    const _Float16* __restrict__ a1p = A + (size_t)(lane15 + 16) * K_DIM + kbase;
    const _Float16* __restrict__ s0r = scales + (size_t)n0 * (K_DIM / GROUP_Q) + kbase / GROUP_Q;
    const _Float16* __restrict__ s1r = scales + (size_t)n1 * (K_DIM / GROUP_Q) + kbase / GROUP_Q;
    const _Float16* __restrict__ z0r = zeros  + (size_t)n0 * (K_DIM / GROUP_Q) + kbase / GROUP_Q;
    const _Float16* __restrict__ z1r = zeros  + (size_t)n1 * (K_DIM / GROUP_Q) + kbase / GROUP_Q;

    v8f acc00 = {}, acc01 = {}, acc10 = {}, acc11 = {};

    // ---- software pipeline: chunk 0 + group 0 scales in flight ----
    Chunk cur = load_chunk(q0, q1, a0p, a1p, khalf * 8);
    _Float16 rs0 = s0r[0], rs1 = s1r[0], rz0 = z0r[0], rz1 = z1r[0];
    h2 s0 = {}, s1 = {}, nzs0 = {}, nzs1 = {};

#pragma unroll 4
    for (int t = 0; t < CHUNKS; ++t) {
        if ((t & 3) == 0) {
            // activate this group's scales; start fetching the next group's
            s0 = (h2){rs0, rs0};
            s1 = (h2){rs1, rs1};
            const _Float16 nz0 = (_Float16)(-((float)rz0 * (float)rs0));
            const _Float16 nz1 = (_Float16)(-((float)rz1 * (float)rs1));
            nzs0 = (h2){nz0, nz0};
            nzs1 = (h2){nz1, nz1};
            const int gn = (t >> 2) + 1;
            if (gn < GRP_PER_WAVE) {
                rs0 = s0r[gn]; rs1 = s1r[gn];
                rz0 = z0r[gn]; rz1 = z1r[gn];
            }
            // pull the weight stream 2 groups ahead (global_prefetch_b8)
            __builtin_prefetch(q0 + (gn + 1) * (GROUP_Q / 2) + khalf * 4, 0, 1);
            __builtin_prefetch(q1 + (gn + 1) * (GROUP_Q / 2) + khalf * 4, 0, 1);
        }

        // issue next chunk's loads (wraps on last iter: harmless redundant load)
        const int tn = (t + 1) & (CHUNKS - 1);
        Chunk nxt = load_chunk(q0, q1, a0p, a1p, tn * 32 + khalf * 8);

        // dequantize current chunk's two B tiles
        BFrag b0, b1;
#pragma unroll
        for (int i = 0; i < 4; ++i) {
            b0.p[i]     = dq_pair((unsigned)cur.w0lo[i], s0, nzs0);
            b0.p[4 + i] = dq_pair((unsigned)cur.w0hi[i], s0, nzs0);
            b1.p[i]     = dq_pair((unsigned)cur.w1lo[i], s1, nzs1);
            b1.p[4 + i] = dq_pair((unsigned)cur.w1hi[i], s1, nzs1);
        }
        AFrag a0, a1;
        a0.h[0] = cur.a0lo; a0.h[1] = cur.a0hi;
        a1.h[0] = cur.a1lo; a1.h[1] = cur.a1hi;

        acc00 = __builtin_amdgcn_wmma_f32_16x16x32_f16(
            false, a0.v, false, b0.v, (short)0, acc00, false, false);
        acc10 = __builtin_amdgcn_wmma_f32_16x16x32_f16(
            false, a1.v, false, b0.v, (short)0, acc10, false, false);
        acc01 = __builtin_amdgcn_wmma_f32_16x16x32_f16(
            false, a0.v, false, b1.v, (short)0, acc01, false, false);
        acc11 = __builtin_amdgcn_wmma_f32_16x16x32_f16(
            false, a1.v, false, b1.v, (short)0, acc11, false, false);

        cur = nxt;
    }

    // ---- K-split reduction through LDS ----
    // C/D layout: VGPR r -> M = r + 8*khalf, N = lane%16 (+16 for tile 1).
#pragma unroll
    for (int r = 0; r < 8; ++r) {
        const int mr = r + khalf * 8;
        red[wave][mr]     [lane15]      = acc00[r];
        red[wave][mr + 16][lane15]      = acc10[r];
        red[wave][mr]     [lane15 + 16] = acc01[r];
        red[wave][mr + 16][lane15 + 16] = acc11[r];
    }
    __syncthreads();

    const int tid = threadIdx.x;
#pragma unroll
    for (int i = 0; i < 8; ++i) {
        const int idx = tid * 8 + i;           // 0..1023 = 32 rows x 32 cols
        const int mm  = idx >> 5;
        const int cc  = idx & 31;
        const float v = red[0][mm][cc] + red[1][mm][cc] +
                        red[2][mm][cc] + red[3][mm][cc];
        const int nc = col0 + cc;
        out[(size_t)mm * N_DIM + nc] = (_Float16)(v + (float)bias[nc]);
    }
}

extern "C" void kernel_launch(void* const* d_in, const int* in_sizes, int n_in,
                              void* d_out, int out_size, void* d_ws, size_t ws_size,
                              hipStream_t stream) {
    (void)in_sizes; (void)n_in; (void)d_ws; (void)ws_size; (void)out_size;
    const _Float16* A       = (const _Float16*)d_in[0];
    const int*      qweight = (const int*)d_in[1];
    const _Float16* scales  = (const _Float16*)d_in[2];
    const _Float16* zeros   = (const _Float16*)d_in[3];
    const _Float16* bias    = (const _Float16*)d_in[4];
    _Float16*       out     = (_Float16*)d_out;

    dim3 grid(N_DIM / COLS);          // 896 blocks
    dim3 block(32 * WAVES);           // 128 threads = 4 wave32
    gptq_w4a16_wmma<<<grid, block, 0, stream>>>(A, qweight, scales, zeros, bias, out);
}
